// WindowAttention_64201171141262
// MI455X (gfx1250) — matrix operand order repaired and checked by reading, not compiled, not run on hardware
//
#include <hip/hip_runtime.h>
#include <hip/hip_bf16.h>

typedef _Float16 half_t;
typedef __attribute__((ext_vector_type(16))) _Float16 v16h;
typedef __attribute__((ext_vector_type(8)))  _Float16 v8h;
typedef __attribute__((ext_vector_type(2)))  _Float16 v2h;
typedef __attribute__((ext_vector_type(8)))  float    v8f;
typedef __attribute__((ext_vector_type(4)))  float    v4f;
typedef __attribute__((ext_vector_type(4)))  int      v4i;

#define WMMA_F16(a, b, c) \
  __builtin_amdgcn_wmma_f32_16x16x32_f16(false, (a), false, (b), (short)0, (c), false, false)

// ---------------------------------------------------------------------------
// CDNA5 async global->LDS copy path (ASYNCcnt), with synchronous fallback.
// Builtin signature (from hipcc diagnostic): (as1 v4i*, as3 v4i*, imm, imm)
// ---------------------------------------------------------------------------
#if defined(__gfx1250__) && \
    __has_builtin(__builtin_amdgcn_global_load_async_to_lds_b128) && \
    __has_builtin(__builtin_amdgcn_s_wait_asynccnt)
#define HAVE_ASYNC_LDS 1
typedef __attribute__((address_space(1))) v4i glb_v4i;
typedef __attribute__((address_space(3))) v4i lds_v4i;
__device__ __forceinline__ void async_copy16(const half_t* g, half_t* l) {
  __builtin_amdgcn_global_load_async_to_lds_b128((glb_v4i*)g, (lds_v4i*)l, 0, 0);
}
#define ASYNC_WAIT() __builtin_amdgcn_s_wait_asynccnt(0)
#else
#define HAVE_ASYNC_LDS 0
#define ASYNC_WAIT()
#endif

// Problem constants
constexpr int BATCH = 32, HH = 56, WW = 56, DIMC = 512, NQKV = 1536;
constexpr int DISP = 3, HEADS = 16, HD = 32;
constexpr int NWH = 8, NWW = 8, WP = 49;                  // 7x7 windows, 8x8 grid
constexpr long long MROWS = (long long)BATCH * HH * WW;   // 100352

// ---------------------------------------------------------------------------
// WMMA fragment loaders (wave32; A and B share a contiguous k order)
// A (16x32 f16): lane<16 -> row=lane,    halves K 0..15 (contiguous)
//                lane>=16-> row=lane-16, halves K 16..31
// B (32x16 f16): lane<16 -> col=lane,    halves K 0..15 (contiguous)
//                lane>=16-> col=lane-16, halves K 16..31
// C/D (16x16 f32): elem v -> row = (lane<16?0:8)+v, col = lane&15
// ---------------------------------------------------------------------------
__device__ __forceinline__ v16h frag_a(const half_t* base, int row, int ld, int lane) {
  int r  = row + (lane & 15);
  int k0 = (lane & 16) ? 16 : 0;
  const half_t* p = base + r * ld + k0;
  v16h out;
  ((v8h*)&out)[0] = *(const v8h*)(p);
  ((v8h*)&out)[1] = *(const v8h*)(p + 8);
  return out;
}
__device__ __forceinline__ v16h frag_b(const half_t* base, int col, int ld, int lane) {
  int c  = col + (lane & 15);
  int k0 = (lane & 16) ? 16 : 0;
  return *(const v16h*)(base + c * ld + k0);
}

// ---------------------------------------------------------------------------
// Kernel 1: qkv = roll(x,-3,-3) @ w_qkv  (fp32 in -> f16 out, f32 acc)
// grid (784, 12), block 256 (8 waves); per-wave 32x64, BK=32,
// double-buffered LDS, one barrier per K step.
// ---------------------------------------------------------------------------
__global__ __launch_bounds__(256) void qkv_gemm(const float* __restrict__ x,
                                                const float* __restrict__ wqkv,
                                                half_t* __restrict__ qkv) {
  __shared__ half_t As[2][128 * 40];  // [m][k], padded
  __shared__ half_t Bs[2][128 * 40];  // [n][k], padded (K-major for B frags)
  const int tid  = threadIdx.x;
  const int lane = tid & 31;
  const int wave = tid >> 5;
  const int bm = blockIdx.x * 128;
  const int bn = blockIdx.y * 128;
  const int wm = (wave >> 1) * 32;
  const int wn = (wave & 1) * 64;

  // A-load assignments (2 per thread), roll fused into the row address
  int arow[2], acol[2];
  long long agrow[2];
#pragma unroll
  for (int pp = 0; pp < 2; ++pp) {
    int idx = tid + pp * 256;
    arow[pp] = idx >> 2;
    acol[pp] = (idx & 3) * 8;
    int m = bm + arow[pp];
    int b = m / (HH * WW);
    int rem = m - b * (HH * WW);
    int h = rem / WW, w = rem - (rem / WW) * WW;
    int hs = h + DISP; if (hs >= HH) hs -= HH;
    int ws2 = w + DISP; if (ws2 >= WW) ws2 -= WW;
    agrow[pp] = ((long long)(b * HH + hs) * WW + ws2) * DIMC;
  }
  const int bkk = (tid >> 4) * 2;   // even k within tile
  const int bnn = (tid & 15) * 8;   // n group of 8

  // register staging for the in-flight tile
  v4f af0[2], af1[2], bq0, bq1, br0, br1;
  auto load_stage = [&](int k0) {
#pragma unroll
    for (int pp = 0; pp < 2; ++pp) {
      const float* src = x + agrow[pp] + k0 + acol[pp];
      af0[pp] = *(const v4f*)(src);
      af1[pp] = *(const v4f*)(src + 4);
    }
    const float* r0p = wqkv + (long long)(k0 + bkk) * NQKV + bn + bnn;
    const float* r1p = r0p + NQKV;
    bq0 = *(const v4f*)(r0p); bq1 = *(const v4f*)(r0p + 4);
    br0 = *(const v4f*)(r1p); br1 = *(const v4f*)(r1p + 4);
  };
  auto store_stage = [&](int buf) {
#pragma unroll
    for (int pp = 0; pp < 2; ++pp) {
      v8h h8;
#pragma unroll
      for (int i = 0; i < 4; ++i) { h8[i] = (half_t)af0[pp][i]; h8[i + 4] = (half_t)af1[pp][i]; }
      *(v8h*)(&As[buf][0] + arow[pp] * 40 + acol[pp]) = h8;
    }
#pragma unroll
    for (int i = 0; i < 8; ++i) {
      float fa = (i < 4) ? bq0[i] : bq1[i - 4];
      float fb = (i < 4) ? br0[i] : br1[i - 4];
      v2h pk = { (half_t)fa, (half_t)fb };
      *(v2h*)(&Bs[buf][0] + (bnn + i) * 40 + bkk) = pk;
    }
  };

  load_stage(0);
  store_stage(0);

  v8f acc[2][4] = {};
  int pbuf = 0;
  for (int k0 = 0; k0 < DIMC; k0 += 32) {
    __syncthreads();                       // buf[pbuf] ready; buf[pbuf^1] free
    const bool more = (k0 + 32 < DIMC);
    if (more) load_stage(k0 + 32);         // overlap HBM with WMMA below
    v16h afr[2], bfr[4];
#pragma unroll
    for (int i = 0; i < 2; ++i) afr[i] = frag_a(&As[pbuf][0], wm + i * 16, 40, lane);
#pragma unroll
    for (int j = 0; j < 4; ++j) bfr[j] = frag_b(&Bs[pbuf][0], wn + j * 16, 40, lane);
#pragma unroll
    for (int i = 0; i < 2; ++i)
#pragma unroll
      for (int j = 0; j < 4; ++j)
        acc[i][j] = WMMA_F16(afr[i], bfr[j], acc[i][j]);
    if (more) store_stage(pbuf ^ 1);
    pbuf ^= 1;
  }

  // epilogue: f16 row-major qkv [M][1536]
  const int r0 = (lane & 16) ? 8 : 0;
  const int cc = lane & 15;
#pragma unroll
  for (int i = 0; i < 2; ++i)
#pragma unroll
    for (int j = 0; j < 4; ++j)
#pragma unroll
      for (int v = 0; v < 8; ++v) {
        int m = bm + wm + i * 16 + r0 + v;
        int n = bn + wn + j * 16 + cc;
        qkv[(long long)m * NQKV + n] = (half_t)acc[i][j][v];
      }
}

// ---------------------------------------------------------------------------
// Kernel 2: per-(batch,head,window) attention, one wave32 per block.
// 49x32 q/k/v padded to 64x32; 16 WMMA for QK^T, 16 WMMA for P*V.
// Q/K rows staged via async global->LDS copies when available.
// grid 32768, block 32
// ---------------------------------------------------------------------------
__global__ __launch_bounds__(32) void win_attn(const half_t* __restrict__ qkv,
                                               const float* __restrict__ pos_emb,
                                               half_t* __restrict__ aout) {
  __shared__ half_t Qs[64 * 40];   // [i][d]
  __shared__ half_t Ks[64 * 40];   // [j][d] == B-frag layout for QK^T
  __shared__ half_t Vt[32 * 72];   // [d][j] == B-frag layout for P*V
  __shared__ half_t Ps[64 * 72];   // [i][j] softmax probs
  __shared__ float  Bias[169];     // pos_emb[.,.,head] (13x13)

  const int lane = threadIdx.x;
  const int wid  = blockIdx.x;
  const int b    = wid >> 10;
  const int head = (wid >> 6) & 15;
  const int win  = wid & 63;
  const int wy = win >> 3, wx = win & 7;

  // zero only the padding rows of Q/K (rows < 49 are written via ASYNCcnt ops,
  // which are unordered w.r.t. DS stores -> must not overlap)
  for (int i = 49 * 40 + lane; i < 64 * 40; i += 32) { Qs[i] = (half_t)0.f; Ks[i] = (half_t)0.f; }
  for (int i = lane; i < 32 * 72; i += 32) Vt[i] = (half_t)0.f;
  for (int i = lane; i < 169; i += 32)     Bias[i] = pos_emb[i * HEADS + head];

  // gather q/k rows; v transposed (manual ds stores)
  for (int p = lane; p < WP; p += 32) {
    int py = p / 7, px = p - (p / 7) * 7;
    int h = wy * 7 + py, w = wx * 7 + px;
    long long grow = (long long)(b * HH + h) * WW + w;
    const half_t* base = qkv + grow * NQKV + head * HD;
#if HAVE_ASYNC_LDS
#pragma unroll
    for (int c = 0; c < 4; ++c) {
      async_copy16(base + c * 8,        Qs + p * 40 + c * 8);
      async_copy16(base + DIMC + c * 8, Ks + p * 40 + c * 8);
    }
#else
    const uint4* q4 = (const uint4*)(base);
    const uint4* k4 = (const uint4*)(base + DIMC);
    uint4* qd = (uint4*)(Qs + p * 40);
    uint4* kd = (uint4*)(Ks + p * 40);
#pragma unroll
    for (int c = 0; c < 4; ++c) { qd[c] = q4[c]; kd[c] = k4[c]; }
#endif
    const uint4* v4 = (const uint4*)(base + 2 * DIMC);
    alignas(16) half_t vr[32];
#pragma unroll
    for (int c = 0; c < 4; ++c) *(uint4*)(vr + c * 8) = v4[c];
#pragma unroll
    for (int d = 0; d < 32; ++d) Vt[d * 72 + p] = vr[d];
  }
  ASYNC_WAIT();
  __syncthreads();

  const float scale = 0.17677669529663687f;   // 1/sqrt(32)
  const bool ul = (wy == NWH - 1);            // last window row -> upper/lower mask
  const bool lr = (wx == NWW - 1);            // last window col -> left/right mask
  const int r0   = (lane & 16) ? 8 : 0;
  const int ccol = lane & 15;

  v16h kb[4];
#pragma unroll
  for (int j = 0; j < 4; ++j) kb[j] = frag_b(Ks, j * 16, 40, lane);

#pragma unroll
  for (int ih = 0; ih < 2; ++ih) {            // two 32-row halves to bound VGPRs
    v8f dot[2][4] = {};
#pragma unroll
    for (int it = 0; it < 2; ++it) {
      v16h qa = frag_a(Qs, (ih * 2 + it) * 16, 40, lane);
#pragma unroll
      for (int jt = 0; jt < 4; ++jt)
        dot[it][jt] = WMMA_F16(qa, kb[jt], dot[it][jt]);
    }
    // bias + shift mask + row softmax (in C-fragment layout)
#pragma unroll
    for (int it = 0; it < 2; ++it) {
#pragma unroll
      for (int v = 0; v < 8; ++v) {
        int i  = (ih * 2 + it) * 16 + r0 + v;
        int ic = (i < WP) ? i : WP - 1;
        int xi = ic / 7, yi = ic - (ic / 7) * 7;
        float rmax = -1e30f;
        float e[4];
#pragma unroll
        for (int jt = 0; jt < 4; ++jt) {
          int j  = jt * 16 + ccol;
          int jc = (j < WP) ? j : WP - 1;
          int xj = jc / 7, yj = jc - (jc / 7) * 7;
          bool masked = (j >= WP) ||
                        (ul && ((xi < 4) != (xj < 4))) ||
                        (lr && ((yi < 4) != (yj < 4)));
          int rel0 = xj - xi + ((xj >= xi) ? 0 : 13);
          int rel1 = yj - yi + ((yj >= yi) ? 0 : 13);
          float d = masked ? -1e30f
                           : dot[it][jt][v] * scale + Bias[rel0 * 13 + rel1];
          e[jt] = d;
          rmax = fmaxf(rmax, d);
        }
#pragma unroll
        for (int s = 8; s >= 1; s >>= 1) rmax = fmaxf(rmax, __shfl_xor(rmax, s, 32));
        float rsum = 0.f;
#pragma unroll
        for (int jt = 0; jt < 4; ++jt) { e[jt] = __expf(e[jt] - rmax); rsum += e[jt]; }
#pragma unroll
        for (int s = 8; s >= 1; s >>= 1) rsum += __shfl_xor(rsum, s, 32);
        float inv = __builtin_amdgcn_rcpf(rsum);
#pragma unroll
        for (int jt = 0; jt < 4; ++jt)
          Ps[i * 72 + jt * 16 + ccol] = (half_t)(e[jt] * inv);
      }
    }
  }
  __syncthreads();

  // out = P (64x64) @ V (64x32): K split in two 32-steps
  v8f oacc[4][2] = {};
#pragma unroll
  for (int s = 0; s < 2; ++s) {
    v16h vb[2];
#pragma unroll
    for (int dt = 0; dt < 2; ++dt) vb[dt] = frag_b(Vt + s * 32, dt * 16, 72, lane);
#pragma unroll
    for (int it = 0; it < 4; ++it) {
      v16h pa = frag_a(Ps + s * 32, it * 16, 72, lane);
#pragma unroll
      for (int dt = 0; dt < 2; ++dt)
        oacc[it][dt] = WMMA_F16(pa, vb[dt], oacc[it][dt]);
    }
  }
  // scatter back to row-major f16 [M][512] (still in shifted frame)
#pragma unroll
  for (int it = 0; it < 4; ++it)
#pragma unroll
    for (int v = 0; v < 8; ++v) {
      int i = it * 16 + r0 + v;
      if (i < WP) {
        int py = i / 7, px = i - (i / 7) * 7;
        int h = wy * 7 + py, w = wx * 7 + px;
        long long grow = (long long)(b * HH + h) * WW + w;
#pragma unroll
        for (int dt = 0; dt < 2; ++dt)
          aout[grow * DIMC + head * HD + dt * 16 + ccol] = (half_t)oacc[it][dt][v];
      }
    }
}

// ---------------------------------------------------------------------------
// Kernel 3: out = attn @ w_out + b_out, roll(+3,+3) fused into the store.
// Double-buffered; A tile (already f16) staged via async global->LDS copies.
// grid (784, 4), block 256
// ---------------------------------------------------------------------------
__global__ __launch_bounds__(256) void out_gemm(const half_t* __restrict__ aout,
                                                const float* __restrict__ wout,
                                                const float* __restrict__ bout,
                                                float* __restrict__ out) {
  __shared__ half_t As[2][128 * 40];
  __shared__ half_t Bs[2][128 * 40];
  const int tid = threadIdx.x, lane = tid & 31, wave = tid >> 5;
  const int bm = blockIdx.x * 128, bn = blockIdx.y * 128;
  const int wm = (wave >> 1) * 32, wn = (wave & 1) * 64;

  const int arow = tid >> 1;
  const int aoff = (tid & 1) * 16;
  const long long abase = (long long)(bm + arow) * DIMC + aoff;
  const int bkk = (tid >> 4) * 2, bnn = (tid & 15) * 8;

#if !HAVE_ASYNC_LDS
  uint4 ar0, ar1;
#endif
  v4f bq0, bq1, br0, br1;

  auto issue_A = [&](int k0, int buf) {
#if HAVE_ASYNC_LDS
    async_copy16(aout + abase + k0,     &As[buf][0] + arow * 40 + aoff);
    async_copy16(aout + abase + k0 + 8, &As[buf][0] + arow * 40 + aoff + 8);
#else
    const uint4* s = (const uint4*)(aout + abase + k0);
    ar0 = s[0]; ar1 = s[1];
#endif
  };
  auto commit_A = [&](int buf) {
#if !HAVE_ASYNC_LDS
    uint4* d = (uint4*)(&As[buf][0] + arow * 40 + aoff);
    d[0] = ar0; d[1] = ar1;
#else
    (void)buf;
#endif
  };
  auto load_B = [&](int k0) {
    const float* r0p = wout + (long long)(k0 + bkk) * DIMC + bn + bnn;
    const float* r1p = r0p + DIMC;
    bq0 = *(const v4f*)(r0p); bq1 = *(const v4f*)(r0p + 4);
    br0 = *(const v4f*)(r1p); br1 = *(const v4f*)(r1p + 4);
  };
  auto store_B = [&](int buf) {
#pragma unroll
    for (int i = 0; i < 8; ++i) {
      float fa = (i < 4) ? bq0[i] : bq1[i - 4];
      float fb = (i < 4) ? br0[i] : br1[i - 4];
      v2h pk = { (half_t)fa, (half_t)fb };
      *(v2h*)(&Bs[buf][0] + (bnn + i) * 40 + bkk) = pk;
    }
  };

  issue_A(0, 0); commit_A(0);
  load_B(0);     store_B(0);
  ASYNC_WAIT();

  v8f acc[2][4] = {};
  int pbuf = 0;
  for (int k0 = 0; k0 < DIMC; k0 += 32) {
    __syncthreads();
    const bool more = (k0 + 32 < DIMC);
    if (more) { issue_A(k0 + 32, pbuf ^ 1); load_B(k0 + 32); }
    v16h afr[2], bfr[4];
#pragma unroll
    for (int i = 0; i < 2; ++i) afr[i] = frag_a(&As[pbuf][0], wm + i * 16, 40, lane);
#pragma unroll
    for (int j = 0; j < 4; ++j) bfr[j] = frag_b(&Bs[pbuf][0], wn + j * 16, 40, lane);
#pragma unroll
    for (int i = 0; i < 2; ++i)
#pragma unroll
      for (int j = 0; j < 4; ++j)
        acc[i][j] = WMMA_F16(afr[i], bfr[j], acc[i][j]);
    if (more) { commit_A(pbuf ^ 1); store_B(pbuf ^ 1); }
    ASYNC_WAIT();            // async A copies into pbuf^1 complete before next barrier
    pbuf ^= 1;
  }

  const int r0 = (lane & 16) ? 8 : 0, cc = lane & 15;
#pragma unroll
  for (int i = 0; i < 2; ++i)
#pragma unroll
    for (int j = 0; j < 4; ++j) {
      int n = bn + wn + j * 16 + cc;
      float bias = bout[n];
#pragma unroll
      for (int v = 0; v < 8; ++v) {
        int m = bm + wm + i * 16 + r0 + v;
        int bb = m / (HH * WW);
        int rem = m - bb * (HH * WW);
        int h = rem / WW, w = rem - (rem / WW) * WW;
        int hs = h + DISP;  if (hs >= HH) hs -= HH;   // roll(+3,+3) on store
        int ws2 = w + DISP; if (ws2 >= WW) ws2 -= WW;
        long long orow = (long long)(bb * HH + hs) * WW + ws2;
        out[orow * DIMC + n] = acc[i][j][v] + bias;
      }
    }
}

// ---------------------------------------------------------------------------
extern "C" void kernel_launch(void* const* d_in, const int* in_sizes, int n_in,
                              void* d_out, int out_size, void* d_ws, size_t ws_size,
                              hipStream_t stream) {
  const float* x       = (const float*)d_in[0];
  const float* w_qkv   = (const float*)d_in[1];
  const float* pos_emb = (const float*)d_in[2];
  const float* w_out   = (const float*)d_in[3];
  const float* b_out   = (const float*)d_in[4];
  float* out = (float*)d_out;

  // workspace: qkv f16 [100352][1536] (308 MB) then attn-out f16 [100352][512] (103 MB)
  half_t* qkv = (half_t*)d_ws;
  half_t* aw  = (half_t*)d_ws + (size_t)MROWS * NQKV;

  qkv_gemm<<<dim3(784, 12), 256, 0, stream>>>(x, w_qkv, qkv);
  win_attn<<<dim3(32768), 32, 0, stream>>>(qkv, pos_emb, aw);
  out_gemm<<<dim3(784, 4), 256, 0, stream>>>(aw, w_out, b_out, out);
}
